// Meta_Fuse_12945031430622
// MI455X (gfx1250) — matrix operand order repaired and checked by reading, not compiled, not run on hardware
//
#include <hip/hip_runtime.h>

// ---------------------------------------------------------------------------
// Meta_Fuse for MI455X (gfx1250): gather-max -> bf16 WMMA GEMMs -> LSTM scans
// GEMMs: v_wmma_f32_16x16x32_bf16, A tile staged via global_load_async_to_lds.
// Scan:  v_pk_fma_f32 packed recurrent matvec.
// ---------------------------------------------------------------------------

typedef __attribute__((ext_vector_type(16))) __bf16 v16bf;
typedef __attribute__((ext_vector_type(8)))  __bf16 v8bf;
typedef __attribute__((ext_vector_type(8)))  float  v8f;
typedef __attribute__((ext_vector_type(2)))  float  f32x2;

union ABfrag { v16bf v; v8bf h[2]; };

static constexpr int NN = 4096;   // sequence / batch dim
static constexpr int HH = 256;    // hidden dim
static constexpr int NH = NN * HH;

// ---------------------------------------------------------------------------
// Gather-max kernels. One thread per (n, h); n constant per 256-thread block
// so each of the 20 gathered rows is a fully coalesced 1 KB read (L2-hot).
// ---------------------------------------------------------------------------
__global__ __launch_bounds__(256) void gather_avg_cat(
    const float* __restrict__ X, const float* __restrict__ Y,
    const int* __restrict__ idx, __bf16* __restrict__ out) {
  int gid = blockIdx.x * 256 + threadIdx.x;
  int n = gid >> 8, h = gid & 255;
  float x0 = -3.4e38f, x1 = -3.4e38f, y2 = -3.4e38f, y3 = -3.4e38f;
  for (int t = 0; t < 20; ++t) {
    const int* ip = idx + ((size_t)t * NN + n) * 4;
    x0 = fmaxf(x0, X[(size_t)ip[0] * HH + h]);
    x1 = fmaxf(x1, X[(size_t)ip[1] * HH + h]);
    y2 = fmaxf(y2, Y[(size_t)ip[2] * HH + h]);
    y3 = fmaxf(y3, Y[(size_t)ip[3] * HH + h]);
  }
  out[(size_t)n * 512 + h]       = (__bf16)(0.5f * (x0 + x1));
  out[(size_t)n * 512 + 256 + h] = (__bf16)(0.5f * (y2 + y3));
}

__global__ __launch_bounds__(256) void gather_cat2(
    const float* __restrict__ S0, const float* __restrict__ S1,
    const float* __restrict__ S2, const float* __restrict__ S3,
    const int* __restrict__ idx,
    __bf16* __restrict__ out1, __bf16* __restrict__ out2) {
  int gid = blockIdx.x * 256 + threadIdx.x;
  int n = gid >> 8, h = gid & 255;
  float a0 = -3.4e38f, a1 = -3.4e38f, a2 = -3.4e38f, a3 = -3.4e38f;
  for (int t = 0; t < 20; ++t) {
    const int* ip = idx + ((size_t)t * NN + n) * 4;
    a0 = fmaxf(a0, S0[(size_t)ip[0] * HH + h]);
    a1 = fmaxf(a1, S1[(size_t)ip[1] * HH + h]);
    a2 = fmaxf(a2, S2[(size_t)ip[2] * HH + h]);
    a3 = fmaxf(a3, S3[(size_t)ip[3] * HH + h]);
  }
  out1[(size_t)n * 512 + h]       = (__bf16)a0;
  out1[(size_t)n * 512 + 256 + h] = (__bf16)a1;
  out2[(size_t)n * 512 + h]       = (__bf16)a2;
  out2[(size_t)n * 512 + 256 + h] = (__bf16)a3;
}

// ---------------------------------------------------------------------------
// bf16 WMMA GEMM: C[4096, Nout] = act(A[4096,K] * W[Nout,K]^T + bias).
// Requires lda == K (true for every call site): the block's 16-row A tile is
// one contiguous 32*K-byte span, staged into LDS once per block with
// global_load_async_to_lds_b128 (ASYNCcnt), then all 8 waves read their A
// fragments from LDS (ds_load_b128) instead of 8x-redundant global loads.
// ---------------------------------------------------------------------------
__global__ __launch_bounds__(256) void gemm_bf16_wmma(
    const __bf16* __restrict__ A, int lda,
    const __bf16* __restrict__ W, int ldw,
    const float* __restrict__ bias,
    float* __restrict__ Cf, __bf16* __restrict__ Cbf,
    int ldc, int K, int relu) {
  __shared__ __align__(16) __bf16 As[16 * 512];   // up to K=512 -> 16 KB

  const int tid   = threadIdx.x;
  const int lane  = tid & 31;
  const int wid   = tid >> 5;
  const int l16   = lane & 15;
  const int khalf = lane >> 4;           // 0 or 1
  const int mbase = blockIdx.x * 16;
  const int nbase = (blockIdx.y * 8 + wid) * 16;

  // --- async-stage the contiguous A tile into LDS -------------------------
  {
    const int nchunks = (16 * K * 2) / 16;             // 16B chunks
    const unsigned lds_base = (unsigned)(size_t)(void*)&As[0];
    const unsigned long long gbase =
        (unsigned long long)(const void*)(A + (size_t)mbase * lda);
    for (int c = tid; c < nchunks; c += 256) {
      unsigned lds_addr = lds_base + (unsigned)c * 16u;
      unsigned long long gaddr = gbase + (unsigned long long)c * 16ull;
      asm volatile("global_load_async_to_lds_b128 %0, %1, off"
                   :: "v"(lds_addr), "v"(gaddr) : "memory");
    }
    asm volatile("s_wait_asynccnt 0" ::: "memory");
  }
  __syncthreads();

  const __bf16* arow = As + (size_t)l16 * K;                 // LDS
  const __bf16* brow = W + (size_t)(nbase + l16) * ldw;      // global (L2)

  v8f acc = {};
  for (int kb = 0; kb < K; kb += 32) {
    ABfrag a, b;
    a.h[0] = *(const v8bf*)(arow + kb + khalf * 8);        // ds_load_b128
    a.h[1] = *(const v8bf*)(arow + kb + 16 + khalf * 8);
    b.h[0] = *(const v8bf*)(brow + kb + khalf * 8);        // global_load_b128
    b.h[1] = *(const v8bf*)(brow + kb + 16 + khalf * 8);
    acc = __builtin_amdgcn_wmma_f32_16x16x32_bf16(
        /*neg_a=*/false, a.v, /*neg_b=*/false, b.v,
        /*c_mod=*/(short)0, acc, /*reuse_a=*/false, /*reuse_b=*/false);
  }

  const int col = nbase + l16;
  float bv = bias ? bias[col] : 0.f;
#pragma unroll
  for (int v = 0; v < 8; ++v) {
    float val = acc[v] + bv;
    if (relu) val = fmaxf(val, 0.f);
    const int row = mbase + v + 8 * khalf;
    if (Cf)  Cf[(size_t)row * ldc + col]  = val;
    if (Cbf) Cbf[(size_t)row * ldc + col] = (__bf16)val;
  }
}

// ---------------------------------------------------------------------------
// Two-layer LSTM recurrence (one workgroup per LSTM, 1024 threads).
// xg already holds x@Wih^T + (bih+bhh). Gates in LDS, cell state in regs.
// Recurrent matvec uses v_pk_fma_f32 (2 MACs/instr) with two independent
// accumulators to break the dependency chain.
// ---------------------------------------------------------------------------
__global__ __launch_bounds__(1024) void lstm_scan(
    const float* __restrict__ xg,    // [4, N, 1024]
    const float* __restrict__ Whh,   // [4, 2, 1024, 256] fp32
    const float* __restrict__ h0,    // [4, 2, 256]
    const float* __restrict__ c0,
    int layer,
    __bf16* __restrict__ hseq) {     // [4, N, 256] bf16
  __shared__ __align__(16) float h_s[HH];
  __shared__ __align__(16) float gates[1024];
  const int i   = blockIdx.x;
  const int tid = threadIdx.x;
  const float* xgi = xg + (size_t)i * NN * 1024;
  const float* Wr  = Whh + (((size_t)i * 2 + layer) * 1024 + tid) * HH;
  float c = 0.f;
  if (tid < HH) {
    h_s[tid] = h0[((size_t)i * 2 + layer) * HH + tid];
    c        = c0[((size_t)i * 2 + layer) * HH + tid];
  }
  __syncthreads();
  for (int t = 0; t < NN; ++t) {
    if (t + 1 < NN) __builtin_prefetch(xgi + (size_t)(t + 1) * 1024 + tid, 0, 1);
    f32x2 acc2a = {xgi[(size_t)t * 1024 + tid], 0.f};
    f32x2 acc2b = {0.f, 0.f};
    const float4* w4 = (const float4*)Wr;
    const float4* h4 = (const float4*)h_s;
#pragma unroll 8
    for (int k = 0; k < HH / 4; ++k) {
      union { float4 f4; f32x2 p[2]; } wu, hu;
      wu.f4 = w4[k];                      // global_load_b128 (L2-hot)
      hu.f4 = h4[k];                      // ds_load_b128
      asm("v_pk_fma_f32 %0, %1, %2, %0"
          : "+v"(acc2a) : "v"(wu.p[0]), "v"(hu.p[0]));
      asm("v_pk_fma_f32 %0, %1, %2, %0"
          : "+v"(acc2b) : "v"(wu.p[1]), "v"(hu.p[1]));
    }
    gates[tid] = acc2a.x + acc2a.y + acc2b.x + acc2b.y;
    __syncthreads();
    if (tid < HH) {
      float ig = 1.f / (1.f + __expf(-gates[tid]));
      float fg = 1.f / (1.f + __expf(-gates[HH + tid]));
      float gg = tanhf(gates[2 * HH + tid]);
      float og = 1.f / (1.f + __expf(-gates[3 * HH + tid]));
      c = fg * c + ig * gg;
      float h = og * tanhf(c);
      h_s[tid] = h;
      hseq[((size_t)i * NN + t) * HH + tid] = (__bf16)h;
    }
    __syncthreads();
  }
}

// ---------------------------------------------------------------------------
// Small elementwise helpers
// ---------------------------------------------------------------------------
__global__ void cvt_bf16(const float* __restrict__ s, __bf16* __restrict__ d, int n) {
  int g = blockIdx.x * 256 + threadIdx.x;
  if (g < n) d[g] = (__bf16)s[g];
}

__global__ void add_vec(const float* __restrict__ a, const float* __restrict__ b,
                        float* __restrict__ d, int n) {
  int g = blockIdx.x * 256 + threadIdx.x;
  if (g < n) d[g] = a[g] + b[g];
}

__global__ void avg2_bf16(const float* __restrict__ a, const float* __restrict__ b,
                          __bf16* __restrict__ d, int n) {
  int g = blockIdx.x * 256 + threadIdx.x;
  if (g < n) d[g] = (__bf16)(0.5f * (a[g] + b[g]));
}

// MF = m_g_f*(cM_g+1) + m_f_f*(cM_f+1) ; DF = d_g_f*(cD_g+1) + d_s_f*(cD_s+1)
__global__ __launch_bounds__(256) void final_combine(
    const float* __restrict__ fc,   // [4, N, H]: m_g_f, d_g_f, m_f_f, d_s_f
    const float* __restrict__ gx,   // [4, N, H]: cM_g, cM_f, cD_g, cD_s
    float* __restrict__ out) {      // [2, N, H]: MF, DF
  int g = blockIdx.x * 256 + threadIdx.x;
  float mg = fc[g], dg = fc[NH + g], mf_ = fc[2 * NH + g], ds = fc[3 * NH + g];
  float cmg = gx[g], cmf = gx[NH + g], cdg = gx[2 * NH + g], cds = gx[3 * NH + g];
  out[g]      = mg * (cmg + 1.f) + mf_ * (cmf + 1.f);
  out[NH + g] = dg * (cdg + 1.f) + ds * (cds + 1.f);
}

// ---------------------------------------------------------------------------
// Launcher
// ---------------------------------------------------------------------------
extern "C" void kernel_launch(void* const* d_in, const int* in_sizes, int n_in,
                              void* d_out, int out_size, void* d_ws, size_t ws_size,
                              hipStream_t stream) {
  const float* mf      = (const float*)d_in[0];
  const float* df      = (const float*)d_in[1];
  const int*   mmdd    = (const int*)d_in[2];
  const int*   ddmm    = (const int*)d_in[3];
  const int*   mdmd    = (const int*)d_in[4];
  const int*   dmdm    = (const int*)d_in[5];
  const float* Wih     = (const float*)d_in[6];   // [4,2,1024,256]
  const float* Whh     = (const float*)d_in[7];   // [4,2,1024,256]
  const float* bih     = (const float*)d_in[8];   // [4,2,1024]
  const float* bhh     = (const float*)d_in[9];
  const float* h0      = (const float*)d_in[10];  // [4,2,256]
  const float* c0      = (const float*)d_in[11];
  const float* fcW     = (const float*)d_in[12];  // [4,256,256]
  const float* fcb     = (const float*)d_in[13];  // [4,256]
  const float* gxW     = (const float*)d_in[14];  // [2,256,256]
  const float* gxb     = (const float*)d_in[15];  // [2,256]
  const float* xxyyW   = (const float*)d_in[16];  // [2,256,512]
  const float* xxyyb   = (const float*)d_in[17];  // [2,256]
  const float* xyxyW   = (const float*)d_in[18];  // [4,256,512]
  const float* xyxyb   = (const float*)d_in[19];  // [4,256]

  char* ws = (char*)d_ws;
  size_t off = 0;
  auto alloc = [&](size_t bytes) -> void* {
    void* p = ws + off;
    off = (off + bytes + 255) & ~(size_t)255;
    return p;
  };

  // bf16 concatenated GEMM inputs [N, 512]
  __bf16* cat_mmdd  = (__bf16*)alloc((size_t)NN * 512 * 2);
  __bf16* cat_ddmm  = (__bf16*)alloc((size_t)NN * 512 * 2);
  __bf16* cat_mdmd1 = (__bf16*)alloc((size_t)NN * 512 * 2);
  __bf16* cat_mdmd2 = (__bf16*)alloc((size_t)NN * 512 * 2);
  __bf16* cat_dmdm1 = (__bf16*)alloc((size_t)NN * 512 * 2);
  __bf16* cat_dmdm2 = (__bf16*)alloc((size_t)NN * 512 * 2);
  // Fusion features (bf16) in LSTM order: mmdd_F, ddmm_F, mdmd_F, dmdm_F
  __bf16* Fbf[4];
  for (int i = 0; i < 4; ++i) Fbf[i] = (__bf16*)alloc((size_t)NH * 2);
  float* tmpA = (float*)alloc((size_t)NH * 4);
  float* tmpB = (float*)alloc((size_t)NH * 4);
  // bf16 weight copies
  __bf16* xxyyWbf = (__bf16*)alloc((size_t)2 * 256 * 512 * 2);
  __bf16* xyxyWbf = (__bf16*)alloc((size_t)4 * 256 * 512 * 2);
  __bf16* WihBf   = (__bf16*)alloc((size_t)4 * 2 * 1024 * 256 * 2);
  __bf16* fcWbf   = (__bf16*)alloc((size_t)4 * 256 * 256 * 2);
  __bf16* gxWbf   = (__bf16*)alloc((size_t)2 * 256 * 256 * 2);
  float*  fusedb  = (float*)alloc((size_t)4 * 2 * 1024 * 4);   // bih + bhh
  // LSTM buffers
  float*  xg   = (float*)alloc((size_t)4 * NN * 1024 * 4);     // gate preacts
  __bf16* hseq = (__bf16*)alloc((size_t)4 * NH * 2);           // layer0 out
  __bf16* yseq = (__bf16*)alloc((size_t)4 * NH * 2);           // layer1 out
  float*  fcout = (float*)alloc((size_t)4 * NH * 4);
  float*  gxout = (float*)alloc((size_t)4 * NH * 4);

  // 1) gather-max fusion inputs
  gather_avg_cat<<<NH / 256, 256, 0, stream>>>(mf, df, mmdd, cat_mmdd);
  gather_avg_cat<<<NH / 256, 256, 0, stream>>>(df, mf, ddmm, cat_ddmm);
  gather_cat2<<<NH / 256, 256, 0, stream>>>(mf, df, mf, df, mdmd, cat_mdmd1, cat_mdmd2);
  gather_cat2<<<NH / 256, 256, 0, stream>>>(df, mf, df, mf, dmdm, cat_dmdm1, cat_dmdm2);

  // 2) weight conversions + fused bias
  auto cvt = [&](const float* s, __bf16* d, int n) {
    cvt_bf16<<<(n + 255) / 256, 256, 0, stream>>>(s, d, n);
  };
  cvt(xxyyW, xxyyWbf, 2 * 256 * 512);
  cvt(xyxyW, xyxyWbf, 4 * 256 * 512);
  cvt(Wih,   WihBf,   4 * 2 * 1024 * 256);
  cvt(fcW,   fcWbf,   4 * 256 * 256);
  cvt(gxW,   gxWbf,   2 * 256 * 256);
  add_vec<<<(8192 + 255) / 256, 256, 0, stream>>>(bih, bhh, fusedb, 8192);

  auto gemm = [&](const __bf16* A, int lda, const __bf16* W, int ldw,
                  const float* bias, float* Cf, __bf16* Cbf,
                  int Nout, int K, int relu) {
    dim3 grid(NN / 16, Nout / 128);
    gemm_bf16_wmma<<<grid, 256, 0, stream>>>(A, lda, W, ldw, bias, Cf, Cbf,
                                             Nout, K, relu);
  };

  // 3) fusion GEMMs -> bf16 features
  gemm(cat_mmdd, 512, xxyyWbf,             512, xxyyb,       nullptr, Fbf[0], 256, 512, 1);
  gemm(cat_ddmm, 512, xxyyWbf + 256 * 512, 512, xxyyb + 256, nullptr, Fbf[1], 256, 512, 1);
  gemm(cat_mdmd1, 512, xyxyWbf,                 512, xyxyb,       tmpA, nullptr, 256, 512, 1);
  gemm(cat_mdmd2, 512, xyxyWbf + 1 * 256 * 512, 512, xyxyb + 256, tmpB, nullptr, 256, 512, 1);
  avg2_bf16<<<(NH + 255) / 256, 256, 0, stream>>>(tmpA, tmpB, Fbf[2], NH);
  gemm(cat_dmdm1, 512, xyxyWbf + 2 * 256 * 512, 512, xyxyb + 512, tmpA, nullptr, 256, 512, 1);
  gemm(cat_dmdm2, 512, xyxyWbf + 3 * 256 * 512, 512, xyxyb + 768, tmpB, nullptr, 256, 512, 1);
  avg2_bf16<<<(NH + 255) / 256, 256, 0, stream>>>(tmpA, tmpB, Fbf[3], NH);

  // 4) LSTM layer 0: xg = F @ Wih0^T + (bih0+bhh0), then scan
  for (int i = 0; i < 4; ++i)
    gemm(Fbf[i], 256, WihBf + ((size_t)i * 2 + 0) * 1024 * 256, 256,
         fusedb + ((size_t)i * 2 + 0) * 1024, xg + (size_t)i * NN * 1024,
         nullptr, 1024, 256, 0);
  lstm_scan<<<4, 1024, 0, stream>>>(xg, Whh, h0, c0, 0, hseq);

  // 5) LSTM layer 1
  for (int i = 0; i < 4; ++i)
    gemm(hseq + (size_t)i * NH, 256, WihBf + ((size_t)i * 2 + 1) * 1024 * 256, 256,
         fusedb + ((size_t)i * 2 + 1) * 1024, xg + (size_t)i * NN * 1024,
         nullptr, 1024, 256, 0);
  lstm_scan<<<4, 1024, 0, stream>>>(xg, Whh, h0, c0, 1, yseq);

  // 6) output FCs
  for (int i = 0; i < 4; ++i)
    gemm(yseq + (size_t)i * NH, 256, fcWbf + (size_t)i * 256 * 256, 256,
         fcb + i * 256, fcout + (size_t)i * NH, nullptr, 256, 256, 0);

  // 7) gx (cancha) projections: order cM_g, cM_f, cD_g, cD_s
  gemm(Fbf[0], 256, gxWbf,             256, gxb,       gxout + 0 * (size_t)NH, nullptr, 256, 256, 0);
  gemm(Fbf[2], 256, gxWbf,             256, gxb,       gxout + 1 * (size_t)NH, nullptr, 256, 256, 0);
  gemm(Fbf[1], 256, gxWbf + 256 * 256, 256, gxb + 256, gxout + 2 * (size_t)NH, nullptr, 256, 256, 0);
  gemm(Fbf[3], 256, gxWbf + 256 * 256, 256, gxb + 256, gxout + 3 * (size_t)NH, nullptr, 256, 256, 0);

  // 8) final residual-gated combine -> d_out = [MF ; DF]
  final_combine<<<NH / 256, 256, 0, stream>>>(fcout, gxout, (float*)d_out);
  (void)in_sizes; (void)n_in; (void)out_size; (void)ws_size;
}
